// GRUModel_Template_30554397344424
// MI455X (gfx1250) — matrix operand order repaired and checked by reading, not compiled, not run on hardware
//
#include <hip/hip_runtime.h>
#include <hip/hip_bf16.h>

// ---------------------------------------------------------------------------
// GRU on gfx1250 (MI455X): WMMA f16 (f32 accum) everywhere.
//   Phase 1: GX[t,b,0:1536] = x[b,t,:] @ W_ih^T + b_ih   (parallel GEMM, f16 out)
//   Phase 2: persistent scan kernel; W_hh slice cached in 192KB LDS per WGP,
//            h state kept in f16 (no cvt on the serial path), grid barrier per
//            timestep, HW tanh/rcp for the gate epilogue.
//   Phase 3: out = h_last @ fc_W^T + fc_b
// ---------------------------------------------------------------------------

typedef __attribute__((ext_vector_type(16))) _Float16 v16h;
typedef __attribute__((ext_vector_type(8)))  _Float16 v8h;
typedef __attribute__((ext_vector_type(8)))  float    v8f;

#define GRU_B   128
#define GRU_T   1024
#define GRU_I   256
#define GRU_H   512
#define GRU_3H  1536
#define GRU_O   128

__device__ __forceinline__ v8f wmma_f16(v16h a, v16h b, v8f c) {
  return __builtin_amdgcn_wmma_f32_16x16x32_f16(false, a, false, b,
                                                (short)0, c, false, false);
}

__device__ __forceinline__ float fast_tanh(float x) {
#if __has_builtin(__builtin_amdgcn_tanhf)
  return __builtin_amdgcn_tanhf(x);       // v_tanh_f32 (TRANS, co-executes w/ WMMA)
#else
  return tanhf(x);
#endif
}

__device__ __forceinline__ float fast_sigmoid(float x) {
  return __builtin_amdgcn_rcpf(1.0f + __expf(-x));   // v_exp_f32 + v_rcp_f32
}

// A fragment (16x32 f16) from row-major f32 source (with cvt).
// Contiguous K order used here (B frags match): lanes 0-15 -> rows 0-15, K in {0..15}; lanes 16-31 K+16.
__device__ __forceinline__ v16h load_a_frag_f32(const float* __restrict__ A,
                                                int lda, int row0, int k0) {
  const int lane = threadIdx.x & 31;
  const int r    = row0 + (lane & 15);
  const int klo  = k0 + ((lane >> 4) << 4);
  const float* p = A + (size_t)r * lda + klo;
  v16h a;
#pragma unroll
  for (int i = 0; i < 8; ++i) a[i]     = (_Float16)p[i];        // K = klo + i
#pragma unroll
  for (int i = 0; i < 8; ++i) a[8 + i] = (_Float16)p[8 + i];    // K = klo+8+i
  return a;
}

// A fragment (16x32 f16) from row-major f16 source (raw b128 loads, no cvt).
__device__ __forceinline__ v16h load_a_frag_f16(const _Float16* __restrict__ A,
                                                int lda, int row0, int k0) {
  const int lane = threadIdx.x & 31;
  const int r    = row0 + (lane & 15);
  const int klo  = k0 + ((lane >> 4) << 4);
  const _Float16* p = A + (size_t)r * lda + klo;
  v8h lo = *(const v8h*)(p);         // K = klo..klo+7      (global_load_b128)
  v8h hi = *(const v8h*)(p + 8);     // K = klo+8..klo+15   (global_load_b128)
  v16h a;
#pragma unroll
  for (int i = 0; i < 8; ++i) { a[i] = lo[i]; a[8 + i] = hi[i]; }
  return a;
}

// B fragment (32x16 f16) where B[k][n] = W[j0+n][k] (W row-major f32, ld=ldw).
__device__ __forceinline__ v16h load_b_frag_wT_f32(const float* __restrict__ W,
                                                   int ldw, int j0, int k0) {
  const int lane = threadIdx.x & 31;
  const int n    = lane & 15;
  const int klo  = k0 + ((lane >> 4) << 4);
  const float* p = W + (size_t)(j0 + n) * ldw + klo;
  v16h b;
#pragma unroll
  for (int i = 0; i < 16; ++i) b[i] = (_Float16)p[i];
  return b;
}

// B fragment from an f16 LDS tile laid out row-major [rows][ldw].
__device__ __forceinline__ v16h load_b_frag_lds(const _Float16* Wl,
                                                int ldw, int r0, int k0) {
  const int lane = threadIdx.x & 31;
  const int n    = lane & 15;
  const int klo  = k0 + ((lane >> 4) << 4);
  const _Float16* p = Wl + (size_t)(r0 + n) * ldw + klo;
  v8h lo = *(const v8h*)(p);        // ds_load_b128
  v8h hi = *(const v8h*)(p + 8);    // ds_load_b128
  v16h b;
#pragma unroll
  for (int i = 0; i < 8; ++i) { b[i] = lo[i]; b[8 + i] = hi[i]; }
  return b;
}

// ---------------------------------------------------------------------------
// Phase 1: GX[t, b, :] = x[b, t, :] @ W_ih^T + b_ih, stored f16.
// grid = (8192, 3), block = 256 (8 waves). Wave computes a 16x64 output tile.
// ---------------------------------------------------------------------------
__global__ void __launch_bounds__(256)
gx_gemm_kernel(const float* __restrict__ x,      // [B, T, I]
               const float* __restrict__ W_ih,   // [3H, I]
               const float* __restrict__ b_ih,   // [3H]
               _Float16* __restrict__ GX)        // [T, B, 3H] f16
{
  const int wave = threadIdx.x >> 5;
  const int lane = threadIdx.x & 31;
  const int row0 = blockIdx.x * 16;                   // flat row = b*T + t
  const int n0   = blockIdx.y * 512 + wave * 64;      // output column base

  const v8f vzero = {};
  v8f c[4];
#pragma unroll
  for (int j = 0; j < 4; ++j) c[j] = vzero;

#pragma unroll 2
  for (int k0 = 0; k0 < GRU_I; k0 += 32) {
    v16h a = load_a_frag_f32(x, GRU_I, row0, k0);
#pragma unroll
    for (int j = 0; j < 4; ++j) {
      v16h bf = load_b_frag_wT_f32(W_ih, GRU_I, n0 + 16 * j, k0);
      c[j] = wmma_f16(a, bf, c[j]);
    }
  }

  // Epilogue: add bias, scatter to GX[t, batch, col] as f16.
  const int batch = row0 >> 10;          // row0 / T
  const int t0    = row0 & (GRU_T - 1);  // row0 % T
  const int N     = lane & 15;
  const int Mb    = (lane >> 4) << 3;
#pragma unroll
  for (int j = 0; j < 4; ++j) {
    const int col  = n0 + 16 * j + N;
    const float bi = b_ih[col];
#pragma unroll
    for (int r = 0; r < 8; ++r) {
      const int t = t0 + Mb + r;
      GX[(size_t)t * (GRU_B * GRU_3H) + (size_t)batch * GRU_3H + col] =
          (_Float16)(c[j][r] + bi);
    }
  }
}

// ---------------------------------------------------------------------------
// Grid-wide barrier (generation counter) for the persistent scan kernel.
// ---------------------------------------------------------------------------
__device__ __forceinline__ void grid_barrier(unsigned* cnt, unsigned* gen,
                                             unsigned nblocks) {
  __syncthreads();
  if (threadIdx.x == 0) {
    __threadfence();
    unsigned g = __hip_atomic_load(gen, __ATOMIC_RELAXED, __HIP_MEMORY_SCOPE_AGENT);
    unsigned arrived =
        __hip_atomic_fetch_add(cnt, 1u, __ATOMIC_ACQ_REL, __HIP_MEMORY_SCOPE_AGENT);
    if (arrived == nblocks - 1u) {
      __hip_atomic_store(cnt, 0u, __ATOMIC_RELAXED, __HIP_MEMORY_SCOPE_AGENT);
      __hip_atomic_store(gen, g + 1u, __ATOMIC_RELEASE, __HIP_MEMORY_SCOPE_AGENT);
    } else {
      while (__hip_atomic_load(gen, __ATOMIC_ACQUIRE, __HIP_MEMORY_SCOPE_AGENT) == g) {
        __builtin_amdgcn_s_sleep(2);
      }
    }
  }
  __syncthreads();
}

// ---------------------------------------------------------------------------
// Phase 2: persistent GRU scan, h state in f16.
// grid = 64 blocks x 128 threads (4 waves). Block (m, nq): batch rows
// 16m..16m+15, h-cols [nq*64, nq*64+64). Each wave owns one 16-col tile for
// all three gates. W_hh slice (3 x 64 x 512 f16 = 192KB) lives in LDS.
// ---------------------------------------------------------------------------
__global__ void __launch_bounds__(128)
gru_scan_kernel(const _Float16* __restrict__ GX,   // [T, B, 3H] f16
                const float* __restrict__ W_hh,    // [3H, H]
                const float* __restrict__ b_hh,    // [3H]
                _Float16* __restrict__ h0,         // ping  [B, H] f16
                _Float16* __restrict__ h1,         // pong  [B, H] f16
                unsigned* __restrict__ sync)       // [0]=counter, [1]=generation
{
  extern __shared__ _Float16 smem[];               // 3 * 64 * 512 f16

  const int m    = blockIdx.x >> 3;                // 0..7  batch tile
  const int nq   = blockIdx.x & 7;                 // 0..7  column quad
  const int wave = threadIdx.x >> 5;               // 0..3
  const int lane = threadIdx.x & 31;
  const int row0 = m * 16;                         // batch rows
  const int n0   = nq * 64 + wave * 16;            // h-column base (0..496)

  // --- Stage W_hh slice into LDS as f16 (once; reused 1024 steps). ---
  const int SLICE = 3 * 64 * GRU_H;                // 98304 elements
  for (int i4 = threadIdx.x; i4 < SLICE / 4; i4 += blockDim.x) {
    const int i = i4 * 4;
    const int g = i >> 15;                         // gate
    const int r = (i >> 9) & 63;                   // local row
    const int k = i & (GRU_H - 1);
    const int grow = g * GRU_H + nq * 64 + r;      // global W_hh row
    const float4 v = *(const float4*)(W_hh + (size_t)grow * GRU_H + k);
    smem[i + 0] = (_Float16)v.x;
    smem[i + 1] = (_Float16)v.y;
    smem[i + 2] = (_Float16)v.z;
    smem[i + 3] = (_Float16)v.w;
  }

  // --- Zero h0 cooperatively across the grid. ---
  for (int i = blockIdx.x * blockDim.x + threadIdx.x; i < GRU_B * GRU_H;
       i += gridDim.x * blockDim.x)
    h0[i] = (_Float16)0.0f;

  grid_barrier(&sync[0], &sync[1], gridDim.x);     // h0 + LDS (via syncthreads) ready

  const int N  = lane & 15;
  const int Mb = (lane >> 4) << 3;
  const int col = n0 + N;
  const float bhr = b_hh[col];
  const float bhz = b_hh[col + GRU_H];
  const float bhn = b_hh[col + 2 * GRU_H];
  const v8f vzero = {};

  for (int t = 0; t < GRU_T; ++t) {
    const _Float16* hc = (t & 1) ? h1 : h0;
    _Float16*       hn = (t & 1) ? h0 : h1;

    v8f cr = vzero, cz = vzero, cn = vzero;
#pragma unroll 4
    for (int k0 = 0; k0 < GRU_H; k0 += 32) {
      v16h a  = load_a_frag_f16(hc, GRU_H, row0, k0);
      v16h br = load_b_frag_lds(smem + 0 * 64 * GRU_H, GRU_H, wave * 16, k0);
      cr = wmma_f16(a, br, cr);
      v16h bz = load_b_frag_lds(smem + 1 * 64 * GRU_H, GRU_H, wave * 16, k0);
      cz = wmma_f16(a, bz, cz);
      v16h bn = load_b_frag_lds(smem + 2 * 64 * GRU_H, GRU_H, wave * 16, k0);
      cn = wmma_f16(a, bn, cn);
    }

    // Fused gate epilogue (HW transcendentals on the serial path).
    const _Float16* gxrow = GX + (size_t)t * (GRU_B * GRU_3H);
#pragma unroll
    for (int r = 0; r < 8; ++r) {
      const int brow = row0 + Mb + r;
      const _Float16* gx = gxrow + (size_t)brow * GRU_3H + col;
      const float gxr = (float)gx[0];
      const float gxz = (float)gx[GRU_H];
      const float gxn = (float)gx[2 * GRU_H];
      const float ghr = cr[r] + bhr;
      const float ghz = cz[r] + bhz;
      const float ghn = cn[r] + bhn;
      const float rg = fast_sigmoid(gxr + ghr);
      const float zg = fast_sigmoid(gxz + ghz);
      const float ng = fast_tanh(gxn + rg * ghn);
      const float hp = (float)hc[(size_t)brow * GRU_H + col];
      hn[(size_t)brow * GRU_H + col] = (_Float16)((1.0f - zg) * ng + zg * hp);
      if (t + 1 < GRU_T)   // pull next step's gx row toward L2/L0
        __builtin_prefetch(gx + (size_t)GRU_B * GRU_3H, 0, 1);
    }

    grid_barrier(&sync[0], &sync[1], gridDim.x);
  }
}

// ---------------------------------------------------------------------------
// Phase 3: out = h_last @ fc_W^T + fc_b.  grid = 8, block = 256 (8 waves).
// ---------------------------------------------------------------------------
__global__ void __launch_bounds__(256)
fc_kernel(const _Float16* __restrict__ h,  // [B, H] f16
          const float* __restrict__ fc_W,  // [O, H]
          const float* __restrict__ fc_b,  // [O]
          float* __restrict__ out)         // [B, O]
{
  const int wave = threadIdx.x >> 5;
  const int lane = threadIdx.x & 31;
  const int row0 = blockIdx.x * 16;
  const int n0   = wave * 16;

  v8f c = {};
#pragma unroll 4
  for (int k0 = 0; k0 < GRU_H; k0 += 32) {
    v16h a  = load_a_frag_f16(h, GRU_H, row0, k0);
    v16h bf = load_b_frag_wT_f32(fc_W, GRU_H, n0, k0);
    c = wmma_f16(a, bf, c);
  }
  const int N  = lane & 15;
  const int Mb = (lane >> 4) << 3;
  const float bias = fc_b[n0 + N];
#pragma unroll
  for (int r = 0; r < 8; ++r)
    out[(size_t)(row0 + Mb + r) * GRU_O + n0 + N] = c[r] + bias;
}

// ---------------------------------------------------------------------------
extern "C" void kernel_launch(void* const* d_in, const int* in_sizes, int n_in,
                              void* d_out, int out_size, void* d_ws, size_t ws_size,
                              hipStream_t stream) {
  const float* x    = (const float*)d_in[0];
  const float* W_ih = (const float*)d_in[1];
  const float* W_hh = (const float*)d_in[2];
  const float* b_ih = (const float*)d_in[3];
  const float* b_hh = (const float*)d_in[4];
  const float* fc_W = (const float*)d_in[5];
  const float* fc_b = (const float*)d_in[6];
  float* out = (float*)d_out;

  // Workspace layout.
  char* ws = (char*)d_ws;
  _Float16* GX = (_Float16*)ws;                                  // 402,653,184 B
  size_t off = (size_t)GRU_T * GRU_B * GRU_3H * sizeof(_Float16);
  _Float16* h0 = (_Float16*)(ws + off); off += (size_t)GRU_B * GRU_H * sizeof(_Float16);
  _Float16* h1 = (_Float16*)(ws + off); off += (size_t)GRU_B * GRU_H * sizeof(_Float16);
  unsigned* sync = (unsigned*)(ws + off);

  hipMemsetAsync(sync, 0, 2 * sizeof(unsigned), stream);

  // Phase 1: input projection GEMM.
  gx_gemm_kernel<<<dim3((GRU_B * GRU_T) / 16, 3), dim3(256), 0, stream>>>(
      x, W_ih, b_ih, GX);

  // Phase 2: persistent recurrence (64 WGP-resident blocks, 192KB LDS each).
  const size_t lds_bytes = (size_t)3 * 64 * GRU_H * sizeof(_Float16);
  gru_scan_kernel<<<dim3(64), dim3(128), lds_bytes, stream>>>(
      GX, W_hh, b_hh, h0, h1, sync);

  // After t = 1023 (odd), the final hidden state lives in h0.
  fc_kernel<<<dim3(GRU_B / 16), dim3(256), 0, stream>>>(h0, fc_W, fc_b, out);
}
